// v8DetectionLoss_16741782520465
// MI455X (gfx1250) — matrix operand order repaired and checked by reading, not compiled, not run on hardware
//
#include <hip/hip_runtime.h>
#include <hip/hip_bf16.h>
#include <math.h>

typedef __attribute__((ext_vector_type(16))) _Float16 v16h;
typedef __attribute__((ext_vector_type(8)))  float    v8f;

#define NC        80
#define A_TOT     8400
#define G_MAX     32
#define TOPK      10
#define WAVES_PER_IMG 525   /* 8400 / 16 anchors-per-wave */
#define WAVES_PER_BLK 8

__device__ __forceinline__ float softplusf(float x) {
  return fmaxf(x, 0.0f) + log1pf(expf(-fabsf(x)));
}

__device__ __forceinline__ void levelOf(int a,
    const float* f0, const float* f1, const float* f2,
    const float** F, int* W, int* HW, int* off, float* stride) {
  if (a < 6400)      { *F = f0; *W = 80; *HW = 6400; *off = 0;    *stride = 8.0f;  }
  else if (a < 8000) { *F = f1; *W = 40; *HW = 1600; *off = 6400; *stride = 16.0f; }
  else               { *F = f2; *W = 20; *HW = 400;  *off = 8000; *stride = 32.0f; }
}

// ---------------------------------------------------------------------------
// k0: zero the scalar accumulators
// acc[0]=sum softplus, acc[1]=cls correction, acc[2]=tss, acc[3]=box, acc[4]=dfl
// ---------------------------------------------------------------------------
__global__ void k0_init(float* acc) {
  if (threadIdx.x < 8) acc[threadIdx.x] = 0.0f;
}

// ---------------------------------------------------------------------------
// k1: streaming decode.
// One wave32 = 16 anchors (lane a and lane a+16 cooperate on anchor a).
//  * DFL: lane computes softmax for 2 sides (side=half, side=2+half),
//    packs probs as the f16 B-matrix fragment of V_WMMA_F32_16X16X32_F16,
//    A = constant projection matrix (row s selects side s, value = bin index).
//    D[s][a] = dist(side s, anchor a) lands in lane a, acc element s.
//  * BCE softplus sum over 80 classes (target-independent part of loss_cls).
//  * Align metric + validity bitmask for all 32 GTs.
// ---------------------------------------------------------------------------
__global__ __launch_bounds__(256)
void k1_decode(const float* __restrict__ feat0, const float* __restrict__ feat1,
               const float* __restrict__ feat2, const int* __restrict__ gt_labels,
               const float* __restrict__ gt_bboxes,
               float* __restrict__ bboxOut, float* __restrict__ alignM,
               unsigned* __restrict__ validM, float* __restrict__ acc, int nWaves) {
  const int lane  = threadIdx.x & 31;
  const int gwave = blockIdx.x * WAVES_PER_BLK + (threadIdx.x >> 5);
  if (gwave >= nWaves) return;                 // wave-uniform: EXEC stays all-ones

  const int bimg  = gwave / WAVES_PER_IMG;
  const int chunk = gwave - bimg * WAVES_PER_IMG;
  const int a0    = chunk * 16;
  const int al    = lane & 15;                 // anchor-in-chunk / matrix column
  const int half  = lane >> 4;                 // which K-half of B this lane feeds
  const int aL    = a0 + al;

  const float* F; int W, HW, off; float stride;
  levelOf(a0, feat0, feat1, feat2, &F, &W, &HW, &off, &stride);
  const int hw = aL - off;
  const float* Fb = F + (size_t)bimg * 144 * HW + hw;   // channel c at Fb[c*HW]
  const float ax = (float)(hw % W) + 0.5f;
  const float ay = (float)(hw / W) + 0.5f;

  // ---- DFL logits: sides {half, 2+half}, 16 bins each -----------------------
  float xa[16], xb[16];
  #pragma unroll
  for (int j = 0; j < 16; ++j) xa[j] = Fb[(size_t)(half * 16 + j) * HW];
  #pragma unroll
  for (int j = 0; j < 16; ++j) xb[j] = Fb[(size_t)(32 + half * 16 + j) * HW];

  float m1 = xa[0], m2 = xb[0];
  #pragma unroll
  for (int j = 1; j < 16; ++j) { m1 = fmaxf(m1, xa[j]); m2 = fmaxf(m2, xb[j]); }
  float s1 = 0.0f, s2 = 0.0f;
  #pragma unroll
  for (int j = 0; j < 16; ++j) { xa[j] = expf(xa[j] - m1); s1 += xa[j];
                                 xb[j] = expf(xb[j] - m2); s2 += xb[j]; }
  const float r1 = 1.0f / s1, r2 = 1.0f / s2;

  v16h bf1, bf2;                 // B fragments: lane holds one column's probs
  #pragma unroll
  for (int j = 0; j < 16; ++j) { bf1[j] = (_Float16)(xa[j] * r1);
                                 bf2[j] = (_Float16)(xb[j] * r2); }

  // ---- A fragments: projection matrix, row s -> side s, value = bin index --
  // A k order used here (contiguous, same as B): lane m = row m; element e -> K=16*half+e
  v16h af1, af2;
  #pragma unroll
  for (int e = 0; e < 16; ++e) {
    const _Float16 kv = (_Float16)(float)(e);   // = proj[bin]
    const bool lo = (half == 0);
    af1[e] = (al == 0 && lo) ? kv : ((al == 1 && !lo) ? kv : (_Float16)0.0f);
    af2[e] = (al == 2 && lo) ? kv : ((al == 3 && !lo) ? kv : (_Float16)0.0f);
  }

  v8f dacc = {};
  dacc = __builtin_amdgcn_wmma_f32_16x16x32_f16(false, af1, false, bf1,
                                                (short)0, dacc, false, false);
  dacc = __builtin_amdgcn_wmma_f32_16x16x32_f16(false, af2, false, bf2,
                                                (short)0, dacc, false, false);
  // D layout: lane a (0..15) holds D[0..7][a]; broadcast rows 0..3 to both halves
  float e0 = dacc[0], e1 = dacc[1], e2 = dacc[2], e3 = dacc[3];
  const float d0 = __shfl(e0, al, 32);
  const float d1 = __shfl(e1, al, 32);
  const float d2 = __shfl(e2, al, 32);
  const float d3 = __shfl(e3, al, 32);

  const float px1 = ax - d0, py1 = ay - d1, px2 = ax + d2, py2 = ay + d3;
  if (half == 0) {
    float4 bb = make_float4(px1, py1, px2, py2);
    *(float4*)(bboxOut + (size_t)(bimg * A_TOT + aL) * 4) = bb;
  }

  // ---- BCE target-independent part: sum softplus over 80 classes -----------
  float sp = 0.0f;
  for (int c = half; c < NC; c += 2)
    sp += softplusf(Fb[(size_t)(64 + c) * HW]);
  #pragma unroll
  for (int o = 16; o > 0; o >>= 1) sp += __shfl_down(sp, o, 32);
  if (lane == 0) atomicAdd(&acc[0], sp);

  // ---- align metric + validity for 32 GTs (lane handles g = half*16+i) -----
  const float eps = 1e-7f;
  const float axp = ax * stride, ayp = ay * stride;
  const float qx1 = px1 * stride, qy1 = py1 * stride;
  const float qx2 = px2 * stride, qy2 = py2 * stride;
  const float w1 = qx2 - qx1, h1 = qy2 - qy1 + eps;
  unsigned vbits = 0;
  for (int i = 0; i < 16; ++i) {
    const int g  = half * 16 + i;
    const int gi = bimg * G_MAX + g;
    const int lab = gt_labels[gi];
    const float bx = gt_bboxes[gi * 4 + 0], by = gt_bboxes[gi * 4 + 1];
    const float bw = gt_bboxes[gi * 4 + 2], bh = gt_bboxes[gi * 4 + 3];
    const bool gvalid = (lab > -1) && ((bx + by + bw + bh) > 0.0f);
    const float cx = bx * 640.0f, cy = by * 640.0f;
    const float gw = bw * 640.0f, gh = bh * 640.0f;
    const float gx1 = cx - gw * 0.5f, gy1 = cy - gh * 0.5f;
    const float gx2 = cx + gw * 0.5f, gy2 = cy + gh * 0.5f;
    const float dmin = fminf(fminf(axp - gx1, ayp - gy1),
                             fminf(gx2 - axp, gy2 - ayp));
    const bool valid = gvalid && (dmin > 1e-9f);
    float aval = 0.0f;
    if (valid) {
      const float w2 = gx2 - gx1, h2 = gy2 - gy1 + eps;
      float iw = fmaxf(fminf(qx2, gx2) - fmaxf(qx1, gx1), 0.0f);
      float ih = fmaxf(fminf(qy2, gy2) - fmaxf(qy1, gy1), 0.0f);
      const float inter = iw * ih;
      const float uni = w1 * h1 + w2 * h2 - inter + eps;
      const float iou = fmaxf(inter / uni, 0.0f);
      const float xs = Fb[(size_t)(64 + lab) * HW];
      const float s  = 1.0f / (1.0f + expf(-xs));
      const float i2 = iou * iou;
      aval = sqrtf(s) * (i2 * i2 * i2);
      vbits |= (1u << i);
    }
    alignM[(size_t)gi * A_TOT + aL] = aval;
  }
  const unsigned ob = __shfl(vbits, lane ^ 16, 32);
  if (half == 0)
    validM[bimg * A_TOT + aL] = (vbits & 0xFFFFu) | (ob << 16);
}

// ---------------------------------------------------------------------------
// k2: top-10 over A=8400 per (b,g) row. Per-thread register top-10 (bubble
// insertion, fully static indexing), then 10 tournament rounds in LDS.
// ---------------------------------------------------------------------------
__global__ __launch_bounds__(256)
void k2_topk(const float* __restrict__ alignM, int* __restrict__ topkIdx) {
  const int bg  = blockIdx.x;
  const int tid = threadIdx.x;
  const float* row = alignM + (size_t)bg * A_TOT;

  float v[TOPK]; int id[TOPK];
  #pragma unroll
  for (int p = 0; p < TOPK; ++p) { v[p] = -1.0f; id[p] = 0; }
  for (int i = tid; i < A_TOT; i += 256) {
    float cv = row[i]; int ci = i;
    #pragma unroll
    for (int p = 0; p < TOPK; ++p) {
      if (cv > v[p]) { float tv = v[p]; int ti = id[p];
                       v[p] = cv; id[p] = ci; cv = tv; ci = ti; }
    }
  }
  __shared__ float sv[256 * TOPK];
  __shared__ int   si[256 * TOPK];
  __shared__ float rv[256];
  __shared__ int   rs[256];
  #pragma unroll
  for (int p = 0; p < TOPK; ++p) { sv[tid * TOPK + p] = v[p];
                                   si[tid * TOPK + p] = id[p]; }
  for (int k = 0; k < TOPK; ++k) {
    __syncthreads();
    float bv = -2.0f; int bslot = tid * TOPK;
    #pragma unroll
    for (int p = 0; p < TOPK; ++p) {
      const int s = tid * TOPK + p;
      if (sv[s] > bv) { bv = sv[s]; bslot = s; }
    }
    rv[tid] = bv; rs[tid] = bslot;
    __syncthreads();
    for (int sdx = 128; sdx > 0; sdx >>= 1) {
      if (tid < sdx && rv[tid + sdx] > rv[tid]) {
        rv[tid] = rv[tid + sdx]; rs[tid] = rs[tid + sdx];
      }
      __syncthreads();
    }
    if (tid == 0) {
      const int slot = rs[0];
      topkIdx[bg * TOPK + k] = si[slot];
      sv[slot] = -3.0f;           // invalidate winner
    }
  }
}

// ---------------------------------------------------------------------------
// k3: per-anchor assignment resolution + sparse fg losses (CIoU, DFL, cls
// correction, tss). Atomic accumulation — only fg anchors (~few k) touch it.
// ---------------------------------------------------------------------------
__global__ __launch_bounds__(256)
void k3_assign(const float* __restrict__ feat0, const float* __restrict__ feat1,
               const float* __restrict__ feat2, const int* __restrict__ gt_labels,
               const float* __restrict__ gt_bboxes,
               const float* __restrict__ bboxIn, const float* __restrict__ alignM,
               const unsigned* __restrict__ validM, const int* __restrict__ topkIdx,
               float* __restrict__ acc) {
  const int blocksPerImg = (A_TOT + 255) / 256;
  const int bimg = blockIdx.x / blocksPerImg;
  const int ta   = (blockIdx.x % blocksPerImg) * 256 + threadIdx.x;

  __shared__ int   sTop[G_MAX * TOPK];
  __shared__ int   sLab[G_MAX];
  __shared__ float sBox[G_MAX * 4];
  for (int i = threadIdx.x; i < G_MAX * TOPK; i += 256)
    sTop[i] = topkIdx[bimg * G_MAX * TOPK + i];
  if (threadIdx.x < G_MAX) {
    const int gi = bimg * G_MAX + threadIdx.x;
    sLab[threadIdx.x] = gt_labels[gi];
    const float cx = gt_bboxes[gi * 4 + 0] * 640.0f;
    const float cy = gt_bboxes[gi * 4 + 1] * 640.0f;
    const float gw = gt_bboxes[gi * 4 + 2] * 640.0f;
    const float gh = gt_bboxes[gi * 4 + 3] * 640.0f;
    sBox[threadIdx.x * 4 + 0] = cx - gw * 0.5f;
    sBox[threadIdx.x * 4 + 1] = cy - gh * 0.5f;
    sBox[threadIdx.x * 4 + 2] = cx + gw * 0.5f;
    sBox[threadIdx.x * 4 + 3] = cy + gh * 0.5f;
  }
  __syncthreads();
  if (ta >= A_TOT) return;

  const unsigned vb = validM[bimg * A_TOT + ta];
  unsigned mp = 0; float maxA = 0.0f;
  for (int g = 0; g < G_MAX; ++g) {
    const float av = alignM[(size_t)(bimg * G_MAX + g) * A_TOT + ta];
    maxA = fmaxf(maxA, av);
    bool member = false;
    #pragma unroll
    for (int k = 0; k < TOPK; ++k) member |= (sTop[g * TOPK + k] == ta);
    if (member && ((vb >> g) & 1u)) mp |= (1u << g);
  }
  if (mp == 0) return;
  if (__popc(mp) > 1) {                       // multi-GT: keep only global max
    unsigned nm = 0, t = mp;
    while (t) {
      const int g = __ffs(t) - 1; t &= t - 1;
      if (alignM[(size_t)(bimg * G_MAX + g) * A_TOT + ta] == maxA) nm |= (1u << g);
    }
    mp = nm;
    if (mp == 0) return;                      // fg collapsed to zero
  }
  const int tg = __ffs(mp) - 1;               // argmax(mask_pos) = first set g
  float posA = 0.0f;
  { unsigned t = mp;
    while (t) { const int g = __ffs(t) - 1; t &= t - 1;
                posA = fmaxf(posA, alignM[(size_t)(bimg * G_MAX + g) * A_TOT + ta]); } }

  const float* F; int W, HW, off; float stride;
  levelOf(ta, feat0, feat1, feat2, &F, &W, &HW, &off, &stride);
  const int hw = ta - off;
  const float* Fb = F + (size_t)bimg * 144 * HW + hw;
  const float ax = (float)(hw % W) + 0.5f;
  const float ay = (float)(hw / W) + 0.5f;

  const float rstr = 1.0f / stride;
  const float tbx1 = sBox[tg * 4 + 0] * rstr, tby1 = sBox[tg * 4 + 1] * rstr;
  const float tbx2 = sBox[tg * 4 + 2] * rstr, tby2 = sBox[tg * 4 + 3] * rstr;
  const float4 pb = *(const float4*)(bboxIn + (size_t)(bimg * A_TOT + ta) * 4);
  const float weight = posA;

  atomicAdd(&acc[2], posA);                                      // tss
  const float xs = Fb[(size_t)(64 + sLab[tg]) * HW];
  atomicAdd(&acc[1], xs * posA);                                 // cls correction

  // CIoU (grid units)
  const float eps = 1e-7f;
  const float w1 = pb.z - pb.x, h1 = pb.w - pb.y + eps;
  const float w2 = tbx2 - tbx1, h2 = tby2 - tby1 + eps;
  const float iw = fmaxf(fminf(pb.z, tbx2) - fmaxf(pb.x, tbx1), 0.0f);
  const float ih = fmaxf(fminf(pb.w, tby2) - fmaxf(pb.y, tby1), 0.0f);
  const float inter = iw * ih;
  const float uni = w1 * h1 + w2 * h2 - inter + eps;
  const float iou = inter / uni;
  const float cw = fmaxf(pb.z, tbx2) - fminf(pb.x, tbx1);
  const float ch = fmaxf(pb.w, tby2) - fminf(pb.y, tby1);
  const float c2 = cw * cw + ch * ch + eps;
  const float dx = tbx1 + tbx2 - pb.x - pb.z;
  const float dy = tby1 + tby2 - pb.y - pb.w;
  const float rho2 = (dx * dx + dy * dy) * 0.25f;
  const float da = atanf(w2 / h2) - atanf(w1 / h1);
  const float vv = 0.40528473456f * da * da;       // 4/pi^2
  const float alpha = vv / (vv - iou + 1.0f + eps);
  const float ciou = iou - (rho2 / c2 + vv * alpha);
  atomicAdd(&acc[3], (1.0f - ciou) * weight);

  // DFL
  const float tv[4] = { ax - tbx1, ay - tby1, tbx2 - ax, tby2 - ay };
  float df = 0.0f;
  #pragma unroll
  for (int s = 0; s < 4; ++s) {
    const float tgt = fminf(fmaxf(tv[s], 0.0f), 15.0f - 0.01f);
    const int   tl  = (int)tgt;
    const float wl  = (float)(tl + 1) - tgt;
    const float wr  = tgt - (float)tl;
    float mx = -1e30f, xl = 0.0f, xr = 0.0f;
    #pragma unroll
    for (int j = 0; j < 16; ++j) {
      const float x = Fb[(size_t)(s * 16 + j) * HW];
      mx = fmaxf(mx, x);
      if (j == tl)     xl = x;
      if (j == tl + 1) xr = x;
    }
    float se = 0.0f;
    #pragma unroll
    for (int j = 0; j < 16; ++j)
      se += expf(Fb[(size_t)(s * 16 + j) * HW] - mx);
    const float lse = mx + logf(se);
    df += -((xl - lse) * wl + (xr - lse) * wr);
  }
  atomicAdd(&acc[4], 0.25f * df * weight);
}

// ---------------------------------------------------------------------------
// k4: final combine
// ---------------------------------------------------------------------------
__global__ void k4_final(const float* __restrict__ acc, float* __restrict__ out,
                         int b) {
  const float tss = fmaxf(acc[2], 1.0f);
  out[0] = (7.5f * acc[3] + 0.5f * (acc[0] - acc[1]) + 1.5f * acc[4]) / tss
           * (float)b;
}

// ---------------------------------------------------------------------------
extern "C" void kernel_launch(void* const* d_in, const int* in_sizes, int n_in,
                              void* d_out, int out_size, void* d_ws, size_t ws_size,
                              hipStream_t stream) {
  const float* f0 = (const float*)d_in[0];
  const float* f1 = (const float*)d_in[1];
  const float* f2 = (const float*)d_in[2];
  const int*   gl = (const int*)d_in[3];
  const float* gb = (const float*)d_in[4];
  const int b = in_sizes[3] / G_MAX;

  char* w = (char*)d_ws;
  auto rnd = [](size_t x) { return (x + 255) & ~(size_t)255; };
  size_t o = 0;
  float*    acc    = (float*)(w + o);    o = rnd(o + 8 * sizeof(float));
  float*    bboxB  = (float*)(w + o);    o = rnd(o + (size_t)b * A_TOT * 4 * sizeof(float));
  float*    alignM = (float*)(w + o);    o = rnd(o + (size_t)b * G_MAX * A_TOT * sizeof(float));
  unsigned* validM = (unsigned*)(w + o); o = rnd(o + (size_t)b * A_TOT * sizeof(unsigned));
  int*      topkI  = (int*)(w + o);      o = rnd(o + (size_t)b * G_MAX * TOPK * sizeof(int));
  (void)ws_size; (void)n_in; (void)out_size;

  k0_init<<<1, 64, 0, stream>>>(acc);

  const int nWaves  = b * WAVES_PER_IMG;
  const int blocks1 = (nWaves + WAVES_PER_BLK - 1) / WAVES_PER_BLK;
  k1_decode<<<blocks1, 256, 0, stream>>>(f0, f1, f2, gl, gb,
                                         bboxB, alignM, validM, acc, nWaves);

  k2_topk<<<b * G_MAX, 256, 0, stream>>>(alignM, topkI);

  const int blocksPerImg = (A_TOT + 255) / 256;
  k3_assign<<<b * blocksPerImg, 256, 0, stream>>>(f0, f1, f2, gl, gb,
                                                  bboxB, alignM, validM, topkI, acc);

  k4_final<<<1, 1, 0, stream>>>(acc, (float*)d_out, b);
}
